// GRUModel_83519934038746
// MI455X (gfx1250) — matrix operand order repaired and checked by reading, not compiled, not run on hardware
//
#include <hip/hip_runtime.h>

// ---------------------------------------------------------------------------
// Fully-fused 2-layer GRU scan for MI455X (gfx1250, wave32, WMMA bf16).
// One wave per 16-row batch tile; scans all T=512 steps. f32 hidden state is
// register-resident; only the bf16 A-fragment image round-trips through LDS.
// All matmuls via v_wmma_f32_16x16x32_bf16 (f32 accumulate); gate
// nonlinearities use v_exp_f32 / v_rcp_f32 (TRANS, co-executes with XDL).
// ---------------------------------------------------------------------------

typedef __attribute__((ext_vector_type(16))) __bf16 v16bf;
typedef __attribute__((ext_vector_type(8)))  float  v8f;

static constexpr int T_    = 512;
static constexpr int I_    = 64;
static constexpr int H1_   = 64;
static constexpr int H2_   = 32;
static constexpr int FCN_  = 16;
static constexpr int BTILE = 16;

// LDS layout (dword offsets). Packed B-fragment images: tile = 32 lanes x 8 dwords.
static constexpr int OFF_WIH1 = 0;                    // 2 KT x 12 NT tiles = 6144 dw
static constexpr int OFF_WHH1 = 6144;                 // 6144 dw
static constexpr int OFF_WIH2 = 12288;                // 2 x 6 = 3072 dw
static constexpr int OFF_WHH2 = 15360;                // 1 x 6 = 1536 dw
static constexpr int OFF_HB1  = 16896;                // bf16[16*64]  -> 512 dw
static constexpr int OFF_HB2  = 17408;                // bf16[16*32]  -> 256 dw
static constexpr int OFF_HB2F = 17664;                // f32 [16*32]  -> 512 dw (epilogue)
static constexpr int SMEM_DW  = 18176;                // 72704 bytes

static __device__ inline unsigned packbf2(float a, float b) {
  union { __bf16 h; unsigned short u; } x, y;
  x.h = (__bf16)a; y.h = (__bf16)b;
  return (unsigned)x.u | ((unsigned)y.u << 16);
}

// Pack W^T (W stored [N][K] row-major, i.e. gate-major) into the wave32 B-fragment
// image for 16x16x32 bf16 WMMA: lane = (half<<4)|n ; dword v holds K = kt*32+8*half+2v (v<4), kt*32+16+8*half+2(v-4) (v>=4).
static __device__ void pack_slab(const float* __restrict__ W, int K, int KT, int NT,
                                 unsigned* dst, int lane) {
  const int nn = lane & 15, half = lane >> 4;
  for (int kt = 0; kt < KT; ++kt)
    for (int nt = 0; nt < NT; ++nt) {
      unsigned* p = dst + (((kt * NT + nt) * 32 + lane) << 3);
      const float* w = W + (size_t)(nt * 16 + nn) * K + kt * 32 + 8 * half;
#pragma unroll
      for (int v = 0; v < 8; ++v) {
        float2 e = *(const float2*)(w + 2 * v + (v < 4 ? 0 : 8));
        p[v] = packbf2(e.x, e.y);
      }
    }
}

static __device__ inline v16bf fragB(const unsigned* smem, int base, int kt, int nt,
                                     int NT, int lane) {
  return *reinterpret_cast<const v16bf*>(smem + base + (((kt * NT + nt) * 32 + lane) << 3));
}

// A-fragment (16x32 bf16) from an f32 row: e[j]=A[m][kt+8*half+j], e[8+j]=A[m][kt+16+8*half+j]
static __device__ inline v16bf frag_a_f32(const float* __restrict__ row, int kt, int half) {
  const float4 q0 = *(const float4*)(row + kt + 8 * half);
  const float4 q1 = *(const float4*)(row + kt + 8 * half + 4);
  const float4 q2 = *(const float4*)(row + kt + 16 + 8 * half);
  const float4 q3 = *(const float4*)(row + kt + 16 + 8 * half + 4);
  v16bf a;
  a[0]=(__bf16)q0.x; a[1]=(__bf16)q0.y; a[2]=(__bf16)q0.z; a[3]=(__bf16)q0.w;
  a[4]=(__bf16)q1.x; a[5]=(__bf16)q1.y; a[6]=(__bf16)q1.z; a[7]=(__bf16)q1.w;
  a[8]=(__bf16)q2.x; a[9]=(__bf16)q2.y; a[10]=(__bf16)q2.z; a[11]=(__bf16)q2.w;
  a[12]=(__bf16)q3.x; a[13]=(__bf16)q3.y; a[14]=(__bf16)q3.z; a[15]=(__bf16)q3.w;
  return a;
}

static __device__ inline v16bf frag_a_bf16(const __bf16* __restrict__ row, int kt, int half) {
  v16bf a;
#pragma unroll
  for (int j = 0; j < 8; ++j) {
    a[j]     = row[kt + 8 * half + j];
    a[8 + j] = row[kt + 16 + 8 * half + j];
  }
  return a;
}

static __device__ inline v16bf zero_frag() {
  v16bf a;
#pragma unroll
  for (int j = 0; j < 16; ++j) a[j] = (__bf16)0.f;
  return a;
}

static __device__ inline v8f splat8(float v) {
  v8f r = {v, v, v, v, v, v, v, v};
  return r;
}

static __device__ inline v8f wmma_bf16(v16bf a, v16bf b, v8f c) {
  return __builtin_amdgcn_wmma_f32_16x16x32_bf16(false, a, false, b, (short)0, c,
                                                 false, false);
}

// Fast gate nonlinearities: v_exp_f32 + v_rcp_f32 (no IEEE division expansion).
static __device__ inline float fast_sigmoid(float v) {
  return __builtin_amdgcn_rcpf(1.f + __expf(-v));
}
static __device__ inline float fast_tanh(float v) {
  float e = __expf(-2.f * v);
  return (1.f - e) * __builtin_amdgcn_rcpf(1.f + e);
}

__global__ __launch_bounds__(32)
void gru2_scan_kernel(const float* __restrict__ x,
                      const float* __restrict__ W_ih1, const float* __restrict__ W_hh1,
                      const float* __restrict__ b_ih1, const float* __restrict__ b_hh1,
                      const float* __restrict__ W_ih2, const float* __restrict__ W_hh2,
                      const float* __restrict__ b_ih2, const float* __restrict__ b_hh2,
                      const float* __restrict__ W_fc,  const float* __restrict__ b_fc,
                      const float* __restrict__ W_out, const float* __restrict__ b_out,
                      float* __restrict__ out) {
  extern __shared__ unsigned smem[];
  const int lane = threadIdx.x;
  const int nn   = lane & 15;
  const int half = lane >> 4;
  const int m    = nn;                       // A-matrix row owned by this lane
  const int b0   = blockIdx.x * BTILE;

  __bf16* hb1  = (__bf16*)(smem + OFF_HB1);
  __bf16* hb2  = (__bf16*)(smem + OFF_HB2);
  float*  hb2f = (float*)(smem + OFF_HB2F);

  // --- one-time: pack weights to bf16 B-fragment images in LDS -------------
  pack_slab(W_ih1, 64, 2, 12, smem + OFF_WIH1, lane);
  pack_slab(W_hh1, 64, 2, 12, smem + OFF_WHH1, lane);
  pack_slab(W_ih2, 64, 2, 6,  smem + OFF_WIH2, lane);
  pack_slab(W_hh2, 32, 1, 6,  smem + OFF_WHH2, lane);

  for (int idx = lane; idx < BTILE * H1_; idx += 32) hb1[idx] = (__bf16)0.f;
  for (int idx = lane; idx < BTILE * H2_; idx += 32) hb2[idx] = (__bf16)0.f;

  // --- biases folded into WMMA C initializer (per-lane column) -------------
  float br[4], bz[4], bxn[4], bhn[4];
#pragma unroll
  for (int i = 0; i < 4; ++i) {
    int c = i * 16 + nn;
    br[i]  = b_ih1[c] + b_hh1[c];
    bz[i]  = b_ih1[64 + c] + b_hh1[64 + c];
    bxn[i] = b_ih1[128 + c];
    bhn[i] = b_hh1[128 + c];
  }
  float cr[2], cz[2], cxn[2], chn[2];
#pragma unroll
  for (int i = 0; i < 2; ++i) {
    int c = i * 16 + nn;
    cr[i]  = b_ih2[c] + b_hh2[c];
    cz[i]  = b_ih2[32 + c] + b_hh2[32 + c];
    cxn[i] = b_ih2[64 + c];
    chn[i] = b_hh2[64 + c];
  }

  // --- register-resident f32 hidden state (static lane<->element ownership)
  float h1s[4][8], h2s[2][8];
#pragma unroll
  for (int i = 0; i < 4; ++i)
#pragma unroll
    for (int j = 0; j < 8; ++j) h1s[i][j] = 0.f;
#pragma unroll
  for (int i = 0; i < 2; ++i)
#pragma unroll
    for (int j = 0; j < 8; ++j) h2s[i][j] = 0.f;

  // h1_{t-1} A-fragments carried across iterations (h0 = 0)
  v16bf ah0 = zero_frag();
  v16bf ah1 = zero_frag();

  const float* xbase = x + (size_t)(b0 + m) * T_ * I_;

  __syncthreads();

  // ======================= sequential scan over time =======================
#pragma unroll 1
  for (int t = 0; t < T_; ++t) {
    const float* xr = xbase + (size_t)t * I_;
    __builtin_prefetch(xr + I_, 0, 1);     // global_prefetch_b8: next step's row

    v16bf ax0 = frag_a_f32(xr, 0, half);
    v16bf ax1 = frag_a_f32(xr, 32, half);
    v16bf ag2 = frag_a_bf16(hb2 + m * H2_, 0, half);   // h2_{t-1}

    // ----- layer 1: gates = x@Wih1^T + h1@Whh1^T (per 16-col group) -------
#pragma unroll
    for (int i = 0; i < 4; ++i) {
      v8f ar = splat8(br[i]);
      ar = wmma_bf16(ax0, fragB(smem, OFF_WIH1, 0, i, 12, lane), ar);
      ar = wmma_bf16(ax1, fragB(smem, OFF_WIH1, 1, i, 12, lane), ar);
      ar = wmma_bf16(ah0, fragB(smem, OFF_WHH1, 0, i, 12, lane), ar);
      ar = wmma_bf16(ah1, fragB(smem, OFF_WHH1, 1, i, 12, lane), ar);

      v8f az = splat8(bz[i]);
      az = wmma_bf16(ax0, fragB(smem, OFF_WIH1, 0, 4 + i, 12, lane), az);
      az = wmma_bf16(ax1, fragB(smem, OFF_WIH1, 1, 4 + i, 12, lane), az);
      az = wmma_bf16(ah0, fragB(smem, OFF_WHH1, 0, 4 + i, 12, lane), az);
      az = wmma_bf16(ah1, fragB(smem, OFF_WHH1, 1, 4 + i, 12, lane), az);

      v8f an = splat8(bxn[i]);
      an = wmma_bf16(ax0, fragB(smem, OFF_WIH1, 0, 8 + i, 12, lane), an);
      an = wmma_bf16(ax1, fragB(smem, OFF_WIH1, 1, 8 + i, 12, lane), an);

      v8f ah = splat8(bhn[i]);
      ah = wmma_bf16(ah0, fragB(smem, OFF_WHH1, 0, 8 + i, 12, lane), ah);
      ah = wmma_bf16(ah1, fragB(smem, OFF_WHH1, 1, 8 + i, 12, lane), ah);

#pragma unroll
      for (int j = 0; j < 8; ++j) {
        float rg = fast_sigmoid(ar[j]);
        float zg = fast_sigmoid(az[j]);
        float ng = fast_tanh(an[j] + rg * ah[j]);
        float hv = (1.f - zg) * ng + zg * h1s[i][j];
        h1s[i][j] = hv;
        int row = j + 8 * half;
        int col = i * 16 + nn;
        hb1[row * H1_ + col] = (__bf16)hv;
      }
    }
    __syncthreads();

    // ----- layer 2: gates = h1_t@Wih2^T + h2@Whh2^T -----------------------
    v16bf an0 = frag_a_bf16(hb1 + m * H1_, 0, half);
    v16bf an1 = frag_a_bf16(hb1 + m * H1_, 32, half);
#pragma unroll
    for (int i = 0; i < 2; ++i) {
      v8f ar = splat8(cr[i]);
      ar = wmma_bf16(an0, fragB(smem, OFF_WIH2, 0, i, 6, lane), ar);
      ar = wmma_bf16(an1, fragB(smem, OFF_WIH2, 1, i, 6, lane), ar);
      ar = wmma_bf16(ag2, fragB(smem, OFF_WHH2, 0, i, 6, lane), ar);

      v8f az = splat8(cz[i]);
      az = wmma_bf16(an0, fragB(smem, OFF_WIH2, 0, 2 + i, 6, lane), az);
      az = wmma_bf16(an1, fragB(smem, OFF_WIH2, 1, 2 + i, 6, lane), az);
      az = wmma_bf16(ag2, fragB(smem, OFF_WHH2, 0, 2 + i, 6, lane), az);

      v8f an = splat8(cxn[i]);
      an = wmma_bf16(an0, fragB(smem, OFF_WIH2, 0, 4 + i, 6, lane), an);
      an = wmma_bf16(an1, fragB(smem, OFF_WIH2, 1, 4 + i, 6, lane), an);

      v8f ah = splat8(chn[i]);
      ah = wmma_bf16(ag2, fragB(smem, OFF_WHH2, 0, 4 + i, 6, lane), ah);

#pragma unroll
      for (int j = 0; j < 8; ++j) {
        float rg = fast_sigmoid(ar[j]);
        float zg = fast_sigmoid(az[j]);
        float ng = fast_tanh(an[j] + rg * ah[j]);
        float hv = (1.f - zg) * ng + zg * h2s[i][j];
        h2s[i][j] = hv;
        int row = j + 8 * half;
        int col = i * 16 + nn;
        hb2[row * H2_ + col] = (__bf16)hv;
      }
    }

    // h1_t fragments become h1_{t-1} fragments of the next step (no reload)
    ah0 = an0;
    ah1 = an1;
    __syncthreads();
  }

  // ----- spill final f32 h2 state to LDS for the FC head -------------------
#pragma unroll
  for (int i = 0; i < 2; ++i)
#pragma unroll
    for (int j = 0; j < 8; ++j)
      hb2f[(j + 8 * half) * H2_ + i * 16 + nn] = h2s[i][j];
  __syncthreads();

  // ----- tiny MLP head on last h2 (f32, VALU) ------------------------------
  if (lane < BTILE) {
    float fcv[FCN_];
#pragma unroll
    for (int c = 0; c < FCN_; ++c) {
      float s = b_fc[c];
#pragma unroll
      for (int k = 0; k < H2_; ++k) s += hb2f[lane * H2_ + k] * W_fc[c * H2_ + k];
      fcv[c] = fmaxf(s, 0.f);
    }
    float o = b_out[0];
#pragma unroll
    for (int c = 0; c < FCN_; ++c) o += fcv[c] * W_out[c];
    out[b0 + lane] = o;
  }
}

extern "C" void kernel_launch(void* const* d_in, const int* in_sizes, int n_in,
                              void* d_out, int out_size, void* d_ws, size_t ws_size,
                              hipStream_t stream) {
  (void)n_in; (void)out_size; (void)d_ws; (void)ws_size;
  const float* x     = (const float*)d_in[0];
  const float* W_ih1 = (const float*)d_in[1];
  const float* W_hh1 = (const float*)d_in[2];
  const float* b_ih1 = (const float*)d_in[3];
  const float* b_hh1 = (const float*)d_in[4];
  const float* W_ih2 = (const float*)d_in[5];
  const float* W_hh2 = (const float*)d_in[6];
  const float* b_ih2 = (const float*)d_in[7];
  const float* b_hh2 = (const float*)d_in[8];
  const float* W_fc  = (const float*)d_in[9];
  const float* b_fc  = (const float*)d_in[10];
  const float* W_out = (const float*)d_in[11];
  const float* b_out = (const float*)d_in[12];

  const int Bn = in_sizes[0] / (T_ * I_);          // 2048
  dim3 grid(Bn / BTILE), block(32);
  size_t shmem = (size_t)SMEM_DW * sizeof(unsigned);
  gru2_scan_kernel<<<grid, block, shmem, stream>>>(
      x, W_ih1, W_hh1, b_ih1, b_hh1, W_ih2, W_hh2, b_ih2, b_hh2,
      W_fc, b_fc, W_out, b_out, (float*)d_out);
}
